// URDMU_18622978195473
// MI455X (gfx1250) — compile-verified, matrix operand order checked by reading, not run on hardware
//
#include <hip/hip_runtime.h>

typedef _Float16 half_t;
typedef __attribute__((ext_vector_type(16))) _Float16 v16h;
typedef __attribute__((ext_vector_type(8)))  _Float16 v8h;
typedef __attribute__((ext_vector_type(4)))  _Float16 v4h;
typedef __attribute__((ext_vector_type(8)))  float    v8f;

#define NTOK   16384L   // 8*2048
#define SEQ    2048
#define DMODEL 512

// gfx1250 16-bit WMMA A/B fragment K-map: element i, lane group g=lane>>4
//   K = (i&7) + 8*g + ((i&8)<<1)
// => elements [0..7] are K-contiguous at base 8g, [8..15] at base 16+8g.
// So a fragment = two contiguous 16B chunks when K is the contiguous axis.
__device__ __forceinline__ v16h frag_combine(v8h lo, v8h hi) {
  return __builtin_shufflevector(lo, hi, 0, 1, 2, 3, 4, 5, 6, 7,
                                 8, 9, 10, 11, 12, 13, 14, 15);
}

// ---------------------------------------------------------------------------
// Generic f16 WMMA GEMM: C = act(A(MxK) * BT + bias*bias_scale) [+C32 in-place]
// BT is the weight PRE-TRANSPOSED to [N][K] (K contiguous) so both A and B
// tiles stage into LDS with pure b128 copies and fragments are ds_load_b128.
// Block tile 64x128, 8 waves (2x4), each wave 32x32 (2x2 WMMA tiles).
// Requires M%64==0, N%128==0, K%32==0 (guaranteed by construction here).
// ---------------------------------------------------------------------------
__global__ __launch_bounds__(256) void gemm_f16(
    const half_t* __restrict__ A, const half_t* __restrict__ BT,
    const float* __restrict__ bias, float bias_scale, int act,
    float* C32, half_t* C16, long ldc, int add_flag,
    long M, long N, long K)
{
  __shared__ __align__(16) half_t As[64][40];     // [m][k] 80B rows (16B aligned)
  __shared__ __align__(16) half_t Bs[128][40];    // [n][k] 80B rows

  const int tid  = threadIdx.x;
  const int lane = tid & 31;
  const int wave = tid >> 5;
  const int wm   = wave >> 2;      // 0..1
  const int wn   = wave & 3;       // 0..3
  const int g    = lane >> 4;
  const int ln16 = lane & 15;

  const long bm = (long)blockIdx.y * 64;
  const long bn = (long)blockIdx.x * 128;

  v8f acc[2][2] = {};

  const int arow = tid >> 2;            // 0..63
  const int acol = (tid & 3) * 8;       // 0,8,16,24

  for (long k0 = 0; k0 < K; k0 += 32) {
    // stage A tile (64x32): one b128 per thread
    *(v8h*)&As[arow][acol] = *(const v8h*)(A + (bm + arow) * K + k0 + acol);
    // stage B tile (128 n-rows x 32 k): two b128 per thread, no scatter
#pragma unroll
    for (int p = 0; p < 2; ++p) {
      int i = tid + p * 256;            // 0..511
      int nrow = i >> 2, ncol = (i & 3) * 8;
      *(v8h*)&Bs[nrow][ncol] = *(const v8h*)(BT + (bn + nrow) * K + k0 + ncol);
    }
    if (k0 + 32 < K) {
      __builtin_prefetch(A + (bm + arow) * K + k0 + 32 + acol, 0, 1);
    }
    __syncthreads();

    v16h af[2], bf[2];
#pragma unroll
    for (int mt = 0; mt < 2; ++mt) {
      int row = wm * 32 + mt * 16 + ln16;
      af[mt] = frag_combine(*(const v8h*)&As[row][8 * g],
                            *(const v8h*)&As[row][16 + 8 * g]);
    }
#pragma unroll
    for (int nt = 0; nt < 2; ++nt) {
      int col = wn * 32 + nt * 16 + ln16;
      bf[nt] = frag_combine(*(const v8h*)&Bs[col][8 * g],
                            *(const v8h*)&Bs[col][16 + 8 * g]);
    }
#pragma unroll
    for (int mt = 0; mt < 2; ++mt)
#pragma unroll
      for (int nt = 0; nt < 2; ++nt)
        acc[mt][nt] = __builtin_amdgcn_wmma_f32_16x16x32_f16(
            false, af[mt], false, bf[nt], (short)0, acc[mt][nt], false, false);
    __syncthreads();
  }

  // epilogue
#pragma unroll
  for (int mt = 0; mt < 2; ++mt)
#pragma unroll
    for (int nt = 0; nt < 2; ++nt)
#pragma unroll
      for (int r = 0; r < 8; ++r) {
        long row = bm + wm * 32 + mt * 16 + r + g * 8;
        long col = bn + wn * 32 + nt * 16 + ln16;
        float v = acc[mt][nt][r];
        if (bias) v += bias[col] * bias_scale;
        if (act == 1)      v = fmaxf(v, 0.f);
        else if (act == 2) v = 0.5f * v * (1.f + erff(v * 0.70710678118654752f));
        if (C32) {
          float o = add_flag ? C32[row * ldc + col] : 0.f;
          C32[row * ldc + col] = o + v;
        }
        if (C16) C16[row * ldc + col] = (half_t)v;
      }
}

// ---------------------------------------------------------------------------
// LayerNorm over 512 features -> f16 output
// ---------------------------------------------------------------------------
__global__ __launch_bounds__(128) void ln_kernel(
    const float* __restrict__ h, const float* __restrict__ gamma,
    const float* __restrict__ beta, half_t* __restrict__ out)
{
  long row = blockIdx.x;
  const float* x = h + row * DMODEL;
  __shared__ float red[4];
  int tid = threadIdx.x;

  float vals[4]; float s = 0.f;
#pragma unroll
  for (int i = 0; i < 4; ++i) { vals[i] = x[tid + i * 128]; s += vals[i]; }
#pragma unroll
  for (int o = 16; o >= 1; o >>= 1) s += __shfl_xor(s, o, 32);
  if ((tid & 31) == 0) red[tid >> 5] = s;
  __syncthreads();
  float mean = (red[0] + red[1] + red[2] + red[3]) * (1.f / 512.f);

  float v = 0.f;
#pragma unroll
  for (int i = 0; i < 4; ++i) { float d = vals[i] - mean; v += d * d; }
#pragma unroll
  for (int o = 16; o >= 1; o >>= 1) v += __shfl_xor(v, o, 32);
  __syncthreads();
  if ((tid & 31) == 0) red[tid >> 5] = v;
  __syncthreads();
  float var = (red[0] + red[1] + red[2] + red[3]) * (1.f / 512.f);
  float rstd = rsqrtf(var + 1e-5f);
#pragma unroll
  for (int i = 0; i < 4; ++i) {
    int c = tid + i * 128;
    out[row * DMODEL + c] = (half_t)((vals[i] - mean) * rstd * gamma[c] + beta[c]);
  }
}

// ---------------------------------------------------------------------------
// Per-layer V transpose: vT[(bh*128 + d)*2048 + key] = V[b,key,h,d]
// so attention P*V B-fragments are key-contiguous b128 global loads.
// grid (32 keyblocks, 32 bh), 256 threads; LDS-tiled (scatter paid once).
// ---------------------------------------------------------------------------
__global__ __launch_bounds__(256) void vT_kernel(
    const half_t* __restrict__ qkvt, half_t* __restrict__ vT)
{
  int bh = blockIdx.y;
  int b = bh >> 2, hh = bh & 3;
  int kb = blockIdx.x;                      // 64 keys per block
  __shared__ __align__(16) half_t tile[128][72];  // [dim][key] 144B rows

#pragma unroll
  for (int p = 0; p < 4; ++p) {
    int i = threadIdx.x + p * 256;          // 0..1023
    int key = i >> 4, dcol = (i & 15) * 8;
    v8h v = *(const v8h*)(qkvt + ((long)b * SEQ + kb * 64 + key) * 2048 + 1024 + hh * 128 + dcol);
#pragma unroll
    for (int j = 0; j < 8; ++j) tile[dcol + j][key] = v[j];
  }
  __syncthreads();
#pragma unroll
  for (int p = 0; p < 4; ++p) {
    int i = threadIdx.x + p * 256;          // 0..1023
    int d = i >> 3, kcol = (i & 7) * 8;
    *(v8h*)(vT + ((long)bh * 128 + d) * SEQ + kb * 64 + kcol) = *(const v8h*)&tile[d][kcol];
  }
}

// ---------------------------------------------------------------------------
// Flash attention: grid(32 qblocks, 32 bh), 4 independent waves; wave owns 16
// query rows. All Q/K/V fragments are direct global b128 loads (K slab per
// head is L2/WGP$ resident; V via vT). Only P staging uses (per-wave) LDS.
// No barriers in the main loop.
// ---------------------------------------------------------------------------
__global__ __launch_bounds__(128) void attn_kernel(
    const half_t* __restrict__ qkvt, const half_t* __restrict__ vT,
    half_t* __restrict__ o16)
{
  int bh = blockIdx.y;
  int b = bh >> 2, hh = bh & 3;
  int qblk = blockIdx.x;
  int tid = threadIdx.x, lane = tid & 31, wave = tid >> 5;
  int g = lane >> 4, ln16 = lane & 15;

  __shared__ __align__(16) half_t Pt[4][16][32];

  const float scale = 0.08838834764831845f;  // 128^-0.5
  long rowbase = (long)b * SEQ;
  int q_row = qblk * 64 + wave * 16 + ln16;

  v16h qf[4];
  const half_t* qptr = qkvt + (rowbase + q_row) * 2048 + hh * 128;
#pragma unroll
  for (int kc = 0; kc < 4; ++kc)
    qf[kc] = frag_combine(*(const v8h*)(qptr + kc * 32 + 8 * g),
                          *(const v8h*)(qptr + kc * 32 + 16 + 8 * g));

  // per-lane base pointers for K (dim-contiguous at fixed key) and V (vT rows)
  const half_t* kbase0 = qkvt + (rowbase + ln16) * 2048 + 512 + hh * 128;       // key=ln16
  const half_t* kbase1 = qkvt + (rowbase + 16 + ln16) * 2048 + 512 + hh * 128;  // key=16+ln16

  v8f oacc[8] = {};
  float mrow[8], lrow[8];
#pragma unroll
  for (int r = 0; r < 8; ++r) { mrow[r] = -1e30f; lrow[r] = 0.f; }

  for (int kt = 0; kt < SEQ / 32; ++kt) {
    long koff = (long)kt * 32 * 2048;
    // S = Q K^T (16 x 32), two 16x16 accums; K fragments direct from global
    v8f s0 = {}, s1 = {};
#pragma unroll
    for (int kc = 0; kc < 4; ++kc) {
      v16h b0 = frag_combine(*(const v8h*)(kbase0 + koff + kc * 32 + 8 * g),
                             *(const v8h*)(kbase0 + koff + kc * 32 + 16 + 8 * g));
      v16h b1 = frag_combine(*(const v8h*)(kbase1 + koff + kc * 32 + 8 * g),
                             *(const v8h*)(kbase1 + koff + kc * 32 + 16 + 8 * g));
      s0 = __builtin_amdgcn_wmma_f32_16x16x32_f16(false, qf[kc], false, b0, (short)0, s0, false, false);
      s1 = __builtin_amdgcn_wmma_f32_16x16x32_f16(false, qf[kc], false, b1, (short)0, s1, false, false);
    }

    // online softmax (row = r + 8*g, cols across 16-lane group)
#pragma unroll
    for (int r = 0; r < 8; ++r) {
      float a = s0[r] * scale, c = s1[r] * scale;
      float tm = fmaxf(a, c);
#pragma unroll
      for (int o = 8; o >= 1; o >>= 1) tm = fmaxf(tm, __shfl_xor(tm, o, 32));
      float nm = fmaxf(mrow[r], tm);
      float alpha = __expf(mrow[r] - nm);
      float p0 = __expf(a - nm), p1 = __expf(c - nm);
      float rs = p0 + p1;
#pragma unroll
      for (int o = 8; o >= 1; o >>= 1) rs += __shfl_xor(rs, o, 32);
      lrow[r] = lrow[r] * alpha + rs;
      mrow[r] = nm;
#pragma unroll
      for (int nc = 0; nc < 8; ++nc) oacc[nc][r] *= alpha;
      Pt[wave][r + g * 8][ln16]      = (half_t)p0;
      Pt[wave][r + g * 8][16 + ln16] = (half_t)p1;
    }

    // O += P * V ; V fragments direct from vT (key-contiguous)
    v16h pf = frag_combine(*(const v8h*)&Pt[wave][ln16][8 * g],
                           *(const v8h*)&Pt[wave][ln16][16 + 8 * g]);
#pragma unroll
    for (int nc = 0; nc < 8; ++nc) {
      const half_t* vp = vT + ((long)bh * 128 + nc * 16 + ln16) * SEQ + kt * 32;
      v16h vf = frag_combine(*(const v8h*)(vp + 8 * g),
                             *(const v8h*)(vp + 16 + 8 * g));
      oacc[nc] = __builtin_amdgcn_wmma_f32_16x16x32_f16(false, pf, false, vf, (short)0, oacc[nc], false, false);
    }
  }

  // normalize + write
#pragma unroll
  for (int r = 0; r < 8; ++r) {
    float inv = 1.f / lrow[r];
    long orow = rowbase + qblk * 64 + wave * 16 + r + g * 8;
    half_t* op = o16 + orow * 1024 + hh * 256;
#pragma unroll
    for (int nc = 0; nc < 8; ++nc) op[nc * 16 + ln16] = (half_t)(oacc[nc][r] * inv);
  }
}

// ---------------------------------------------------------------------------
// Column sums of exp(-|i-j|/e) (truncated band +-80, error < 1e-12 rel)
// ---------------------------------------------------------------------------
__global__ void colsum_kernel(float* __restrict__ colsum)
{
  int j = blockIdx.x * blockDim.x + threadIdx.x;
  if (j >= SEQ) return;
  int lo = j - 80 < 0 ? 0 : j - 80;
  int hi = j + 80 > SEQ - 1 ? SEQ - 1 : j + 80;
  float s = 0.f;
  for (int m = lo; m <= hi; ++m)
    s += __expf(-fabsf((float)(j - m)) * 0.36787944117144233f);
  colsum[j] = s;
}

// ---------------------------------------------------------------------------
// o2[i,d] = sum_j exp(-|i-j|/e)/colsum[j] * t[j,d]  (banded, +-80)
// writes o16 feature h*256 + 128 + d; each thread owns 4 contiguous dims (b64)
// ---------------------------------------------------------------------------
__global__ __launch_bounds__(128) void o2_kernel(
    const half_t* __restrict__ qkvt, const float* __restrict__ colsum,
    half_t* __restrict__ o16)
{
  long row = blockIdx.x;
  long b = row >> 11;
  int i = (int)(row & 2047);
  int lo = i - 80 < 0 ? 0 : i - 80;
  int hi = i + 80 > SEQ - 1 ? SEQ - 1 : i + 80;

  __shared__ float wsh[161];
  for (int j = lo + (int)threadIdx.x; j <= hi; j += 128)
    wsh[j - lo] = __expf(-fabsf((float)(i - j)) * 0.36787944117144233f) / colsum[j];
  __syncthreads();

  int f0 = threadIdx.x * 4;  // 4 contiguous t-dims per thread
  float acc[4] = {0.f, 0.f, 0.f, 0.f};
  for (int j = lo; j <= hi; ++j) {
    float w = wsh[j - lo];
    v4h tv = *(const v4h*)(qkvt + ((b << 11) + j) * 2048 + 1536 + f0);
#pragma unroll
    for (int c = 0; c < 4; ++c) acc[c] += w * (float)tv[c];
  }
#pragma unroll
  for (int c = 0; c < 4; ++c) {
    int f = f0 + c;
    int hh = f >> 7, d = f & 127;
    o16[row * 1024 + hh * 256 + 128 + d] = (half_t)acc[c];
  }
}

// ---------------------------------------------------------------------------
// Memory retrieval: s = sigmoid(h.amem^T/sqrt(D)) @ amem + sigmoid(h.nmem^T/..) @ nmem
// ---------------------------------------------------------------------------
__global__ __launch_bounds__(256) void mem_kernel(
    const float* __restrict__ h, const float* __restrict__ amem,
    const float* __restrict__ nmem, half_t* __restrict__ s16)
{
  long row = blockIdx.x;
  __shared__ float hrow[512];
  __shared__ float red[8];
  __shared__ float attsh;
  int tid = threadIdx.x;
  hrow[tid]       = h[row * DMODEL + tid];
  hrow[tid + 256] = h[row * DMODEL + tid + 256];
  __syncthreads();

  const float ms = 0.044194173824159216f;  // 512^-0.5
  float acc0 = 0.f, acc1 = 0.f;
  for (int k = 0; k < 60; ++k) {
    for (int which = 0; which < 2; ++which) {
      const float* mem = which ? nmem : amem;
      float d = hrow[tid] * mem[k * 512 + tid] + hrow[tid + 256] * mem[k * 512 + tid + 256];
#pragma unroll
      for (int o = 16; o >= 1; o >>= 1) d += __shfl_xor(d, o, 32);
      if ((tid & 31) == 0) red[tid >> 5] = d;
      __syncthreads();
      if (tid == 0) {
        float s = 0.f;
        for (int w = 0; w < 8; ++w) s += red[w];
        attsh = 1.f / (1.f + __expf(-s * ms));
      }
      __syncthreads();
      float att = attsh;
      acc0 += att * mem[k * 512 + tid];
      acc1 += att * mem[k * 512 + tid + 256];
      __syncthreads();
    }
  }
  s16[row * DMODEL + tid]       = (half_t)acc0;
  s16[row * DMODEL + tid + 256] = (half_t)acc1;
}

// ---------------------------------------------------------------------------
// classifier head: out[row] = sigmoid(dot(hid[128], w2) + b2)
// ---------------------------------------------------------------------------
__global__ __launch_bounds__(128) void cls2_kernel(
    const half_t* __restrict__ hid, const float* __restrict__ w2,
    const float* __restrict__ b2, float* __restrict__ out)
{
  long row = blockIdx.x;
  int tid = threadIdx.x;
  float v = (float)hid[row * 128 + tid] * w2[tid];
#pragma unroll
  for (int o = 16; o >= 1; o >>= 1) v += __shfl_xor(v, o, 32);
  __shared__ float red[4];
  if ((tid & 31) == 0) red[tid >> 5] = v;
  __syncthreads();
  if (tid == 0)
    out[row] = 1.f / (1.f + __expf(-(red[0] + red[1] + red[2] + red[3] + b2[0])));
}

// ---------------------------------------------------------------------------
// conversions / im2col
// ---------------------------------------------------------------------------
__global__ void f32_to_f16(const float* __restrict__ in, half_t* __restrict__ out, long n)
{
  long i = (long)blockIdx.x * blockDim.x + threadIdx.x;
  if (i < n) out[i] = (half_t)in[i];
}

// transposed f16 conversion: dst[n*K + k] = (f16) src[k*N + n]   (one-time)
__global__ void f32_to_f16_T(const float* __restrict__ src, half_t* __restrict__ dst,
                             long K, long N)
{
  long i = (long)blockIdx.x * blockDim.x + threadIdx.x;
  if (i >= K * N) return;
  long k = i / N, n = i % N;
  dst[n * K + k] = (half_t)src[i];
}

__global__ __launch_bounds__(256) void h_to_z16(const float* __restrict__ h, half_t* __restrict__ z)
{
  long row = blockIdx.x;
  int c = threadIdx.x;
  z[row * 1024 + c]       = (half_t)h[row * DMODEL + c];
  z[row * 1024 + c + 256] = (half_t)h[row * DMODEL + c + 256];
}

__global__ __launch_bounds__(256) void im2col_kernel(const float* __restrict__ x, half_t* __restrict__ out)
{
  long row = blockIdx.x;
  long b = row >> 11;
  int i = (int)(row & 2047);
  for (int c = threadIdx.x; c < 3072; c += 256) {
    int s = c >> 10, cc = c & 1023;
    int src = i + s - 1;
    float v = (src < 0 || src > SEQ - 1) ? 0.f : x[(b * SEQ + src) * 1024 + cc];
    out[row * 3072 + c] = (half_t)v;
  }
}

// ---------------------------------------------------------------------------
// host orchestration
// ---------------------------------------------------------------------------
extern "C" void kernel_launch(void* const* d_in, const int* in_sizes, int n_in,
                              void* d_out, int out_size, void* d_ws, size_t ws_size,
                              hipStream_t stream)
{
  const float* x      = (const float*)d_in[0];
  const float* conv_w = (const float*)d_in[1];
  const float* conv_b = (const float*)d_in[2];
  const float* ln1_g  = (const float*)d_in[3];
  const float* ln1_b  = (const float*)d_in[4];
  const float* qkv_w  = (const float*)d_in[5];
  const float* out_w  = (const float*)d_in[6];
  const float* out_b  = (const float*)d_in[7];
  const float* ln2_g  = (const float*)d_in[8];
  const float* ln2_b  = (const float*)d_in[9];
  const float* ff_w1  = (const float*)d_in[10];
  const float* ff_b1  = (const float*)d_in[11];
  const float* ff_w2  = (const float*)d_in[12];
  const float* ff_b2  = (const float*)d_in[13];
  const float* amem   = (const float*)d_in[14];
  const float* nmem   = (const float*)d_in[15];
  const float* enc_w  = (const float*)d_in[16];
  const float* enc_b  = (const float*)d_in[17];
  const float* cls_w1 = (const float*)d_in[18];
  const float* cls_b1 = (const float*)d_in[19];
  const float* cls_w2 = (const float*)d_in[20];
  const float* cls_b2 = (const float*)d_in[21];
  float* out = (float*)d_out;

  char* ws = (char*)d_ws;
  // workspace layout (bytes)
  constexpr size_t OFF_H      = 0;                       // 16384*512*4   = 32MB  f32 hidden
  constexpr size_t OFF_XN16   = 33554432;                // 16384*512*2   = 16MB  LN out f16
  constexpr size_t OFF_BIG    = 50331648;                // 100.66MB: im2col then qkvt16
  constexpr size_t OFF_O16    = 150994944;               // 16384*1024*2  = 32MB  attn out / z16
  constexpr size_t OFF_TMP16  = 184549376;               // 16384*512*2   = 16MB  ff hidden / s16 / hid16
  constexpr size_t OFF_W16    = 201326592;               // f16 weights (transposed)
  constexpr size_t OFF_CS     = 213647360;               // 2048*4 colsums
  constexpr size_t OFF_VT     = 213655552;               // 32*128*2048*2 = 16MB V transposed

  float*  h      = (float*)(ws + OFF_H);
  half_t* xn16   = (half_t*)(ws + OFF_XN16);
  half_t* big16  = (half_t*)(ws + OFF_BIG);   // im2col (16384x3072) then qkvt (16384x2048)
  half_t* o16    = (half_t*)(ws + OFF_O16);   // attn concat out; later z16 (16384x1024)
  half_t* tmp16  = (half_t*)(ws + OFF_TMP16);
  half_t* wbase  = (half_t*)(ws + OFF_W16);
  float*  csum   = (float*)(ws + OFF_CS);
  half_t* vT     = (half_t*)(ws + OFF_VT);

  // all weights stored TRANSPOSED: [N][K], K contiguous
  half_t* conv_wT = wbase;                 // 512 x 3072
  half_t* qkv_wT  = wbase + 1572864;       // 2 x (2048 x 512)
  half_t* out_wT  = wbase + 3670016;       // 2 x (512 x 1024)
  half_t* ffw1_T  = wbase + 4718592;       // 2 x (512 x 512)
  half_t* ffw2_T  = wbase + 5242880;       // 2 x (512 x 512)
  half_t* enc_wT  = wbase + 5767168;       // 512 x 512
  half_t* clsw1_T = wbase + 6029312;       // 128 x 1024

  auto cvtT = [&](const float* s, half_t* d, long K, long N) {
    long n = K * N;
    f32_to_f16_T<<<dim3((unsigned)((n + 255) / 256)), 256, 0, stream>>>(s, d, K, N);
  };
  auto gemm = [&](const half_t* A, const half_t* BT, const float* bias, float bscale,
                  int act, float* C32, half_t* C16, long ldc, int add,
                  long M, long N, long K) {
    dim3 grid((unsigned)(N / 128), (unsigned)(M / 64));
    gemm_f16<<<grid, 256, 0, stream>>>(A, BT, bias, bscale, act, C32, C16, ldc, add, M, N, K);
  };

  // one-time transposed weight conversion
  cvtT(conv_w, conv_wT, 3072, 512);
  for (int l = 0; l < 2; ++l) {
    cvtT(qkv_w + (long)l * 512 * 2048, qkv_wT + (long)l * 512 * 2048, 512, 2048);
    cvtT(out_w + (long)l * 1024 * 512, out_wT + (long)l * 1024 * 512, 1024, 512);
    cvtT(ff_w1 + (long)l * 512 * 512,  ffw1_T + (long)l * 512 * 512,  512, 512);
    cvtT(ff_w2 + (long)l * 512 * 512,  ffw2_T + (long)l * 512 * 512,  512, 512);
  }
  cvtT(enc_w,  enc_wT,  512, 512);
  cvtT(cls_w1, clsw1_T, 1024, 128);

  // conv1d (k=3) as im2col GEMM, fused bias + ReLU -> h (f32)
  im2col_kernel<<<dim3((unsigned)NTOK), 256, 0, stream>>>(x, big16);
  colsum_kernel<<<dim3(8), 256, 0, stream>>>(csum);
  gemm(big16, conv_wT, conv_b, 1.f, /*relu*/1, h, nullptr, DMODEL, 0,
       NTOK, DMODEL, 3072);

  // transformer layers
  for (int l = 0; l < 2; ++l) {
    ln_kernel<<<dim3((unsigned)NTOK), 128, 0, stream>>>(h, ln1_g + l * 512, ln1_b + l * 512, xn16);
    // qkvt = xn @ qkv_w[l]  (16384x512 x 512x2048) -> f16
    gemm(xn16, qkv_wT + (long)l * 512 * 2048, nullptr, 0.f, 0,
         nullptr, big16, 2048, 0, NTOK, 2048, 512);
    // transpose V once, then flash attention (o1) + banded A2 @ t (o2)
    vT_kernel<<<dim3(32, 32), 256, 0, stream>>>(big16, vT);
    attn_kernel<<<dim3(32, 32), 128, 0, stream>>>(big16, vT, o16);
    o2_kernel<<<dim3((unsigned)NTOK), 128, 0, stream>>>(big16, csum, o16);
    // h += o @ out_w[l] + out_b[l]
    gemm(o16, out_wT + (long)l * 1024 * 512, out_b + l * 512, 1.f, 0,
         h, nullptr, DMODEL, 1, NTOK, DMODEL, 1024);
    // FFN with residual
    ln_kernel<<<dim3((unsigned)NTOK), 128, 0, stream>>>(h, ln2_g + l * 512, ln2_b + l * 512, xn16);
    gemm(xn16, ffw1_T + (long)l * 512 * 512, ff_b1 + l * 512, 1.f, /*gelu*/2,
         nullptr, tmp16, DMODEL, 0, NTOK, DMODEL, 512);
    gemm(tmp16, ffw2_T + (long)l * 512 * 512, ff_b2 + l * 512, 1.f, 0,
         h, nullptr, DMODEL, 1, NTOK, DMODEL, 512);
  }

  // memory retrieval: s16 = sig(h amem^T)amem + sig(h nmem^T)nmem (f16)
  mem_kernel<<<dim3((unsigned)NTOK), 256, 0, stream>>>(h, amem, nmem, tmp16);
  // z = [h | s @ enc_w + 2*enc_b]  (f16, ld 1024, reuses o16)
  h_to_z16<<<dim3((unsigned)NTOK), 256, 0, stream>>>(h, o16);
  gemm(tmp16, enc_wT, enc_b, 2.f, 0, nullptr, o16 + 512, 1024, 0,
       NTOK, DMODEL, 512);
  // classifier: relu(z @ cls_w1 + b1) -> hid16 (reuses tmp16), then sigmoid head
  gemm(o16, clsw1_T, cls_b1, 1.f, /*relu*/1, nullptr, tmp16, 128, 0,
       NTOK, 128, 1024);
  cls2_kernel<<<dim3((unsigned)NTOK), 128, 0, stream>>>(tmp16, cls_w2, cls_b2, out);
}
